// Codec_82111184765048
// MI455X (gfx1250) — matrix operand order repaired and checked by reading, not compiled, not run on hardware
//
#include <hip/hip_runtime.h>
#include <hip/hip_bf16.h>
#include <math.h>
#include <stdint.h>

typedef __attribute__((ext_vector_type(16))) _Float16 v16h;
typedef __attribute__((ext_vector_type(8)))  _Float16 v8h;
typedef __attribute__((ext_vector_type(8)))  float    v8f;

// ---------------- network topology (shared host/device, constexpr) ----------
#define NSTAGES 14
#define MAXL    48
#define NLAYERS 44

struct Net {
    int nlayers;
    int stage_first[NSTAGES];
    int stage_count[NSTAGES];
    int lin[MAXL], lout[MAXL], woff[MAXL], boff[MAXL];
    int wtotal; // floats per channel (weights + biases, packed)
};

constexpr Net make_net() {
    Net n{};
    int off = 0, L = 0;
    for (int s = 0; s < NSTAGES; ++s) {
        int kp = s + 2;
        int nch[8]; int m = 0;
        nch[m++] = 2 * kp - 1;
        int k = kp;
        while (k >= 1) { nch[m++] = k; k >>= 1; }
        n.stage_first[s] = L;
        n.stage_count[s] = m - 1;
        for (int i = 0; i < m - 1; ++i) {
            n.lin[L]  = nch[i];
            n.lout[L] = nch[i + 1];
            n.woff[L] = off; off += nch[i + 1] * nch[i];
            n.boff[L] = off; off += nch[i + 1];
            ++L;
        }
    }
    n.nlayers = L;
    n.wtotal  = off;
    return n;
}
constexpr Net NET = make_net();

// ---------------- problem constants ----------------------------------------
#define IMG       512
#define BATCH     32
#define CHAN      3
#define BC        (BATCH * CHAN)           // 96
#define NBLOCKS   16384                    // (512/4)^2 per (b,c)
#define TILES_BC  (NBLOCKS / 16)           // 1024 tiles of 16 rows
#define CHUNKS    32                       // grid.y
#define TILES_WG  (TILES_BC / CHUNKS)      // 32 tiles per workgroup
#define NWAVES    4
#define TILES_WAVE (TILES_WG / NWAVES)     // 8 tiles per wave
#define TPB       128
#define NX_ELEMS  100663296.0f             // 128*3*512*512
#define ND_ELEMS  23592960.0f              // 32*3*16384*15
#define RES_X     262144.0f                // 16384*16
#define RES_D     245760.0f                // 16384*15

__device__ inline void histAdd(unsigned* h, float v) {
    if (v >= -1.0f && v <= 1.0f) {
        int b = (int)floorf((v + 1.0f) * 128.0f);
        b = b > 255 ? 255 : (b < 0 ? 0 : b);
        atomicAdd(&h[b], 1u);
    }
}

// Async global->LDS 16B copy (ASYNCcnt-tracked; dsaddr = wave-relative LDS
// byte offset = low 32 bits of the generic shared-pointer address).
__device__ inline void asyncCopy16(uint32_t lds_byte, const float* gptr) {
    asm volatile("global_load_async_to_lds_b128 %0, %1, off"
                 :: "v"(lds_byte), "v"(gptr)
                 : "memory");
}
__device__ inline void asyncWait0() {
#if __has_builtin(__builtin_amdgcn_s_wait_asynccnt)
    __builtin_amdgcn_s_wait_asynccnt(0);
#else
    asm volatile("s_wait_asynccnt 0x0" ::: "memory");
#endif
}

// ---------------- main kernel ----------------------------------------------
__global__ __launch_bounds__(TPB)
void codec_main(const float* __restrict__ x, const float* __restrict__ params,
                unsigned* __restrict__ hist_x, unsigned* __restrict__ hist_d,
                float* __restrict__ sums) {
    // Pre-baked B fragments: [layer][lane][16 halfs]; each lane reads its
    // 32 bytes with two ds_load_b128.
    __shared__ alignas(16) _Float16 s_bfrag[NLAYERS][32][16];
    __shared__ float    s_biasL[NLAYERS][16];          // zero-padded biases
    __shared__ alignas(16) float s_xs[2][NWAVES][16][16]; // double-buffered pixels
    __shared__ float    s_e[NWAVES][16][16];           // error chain per tile row
    __shared__ alignas(16) _Float16 s_act[NWAVES][16][32]; // activation tile
    __shared__ unsigned s_hx[256], s_hd[256];
    __shared__ float    s_sq[2];

    const int tid  = threadIdx.x;
    const int wv   = tid >> 5;
    const int lane = tid & 31;
    const int bc   = blockIdx.x;     // b*3 + c
    const int ch   = bc % CHAN;

    const float* xbase = x + (size_t)bc * (IMG * IMG);

    // ---- prefetch tile 0 pixels into buffer 0 (overlaps param bake) ----
    // tile = 16 blocks x 4 image-rows x 16B = 64 chunks; 2 chunks per lane.
    {
        const int tile0 = blockIdx.y * TILES_WG + wv * TILES_WAVE;
        #pragma unroll
        for (int q8 = 0; q8 < 2; ++q8) {
            const int q   = q8 * 32 + lane;
            const int blk = q >> 2, r = q & 3;
            const int blkId = tile0 * 16 + blk;
            const int bi = blkId >> 7, bj = blkId & 127;
            const float* g = xbase + (size_t)(bi * 4 + r) * IMG + (size_t)(bj * 4);
            const uint32_t loff = (uint32_t)(uintptr_t)&s_xs[0][wv][blk][r * 4];
            asyncCopy16(loff, g);
        }
    }

    for (int i = tid; i < 256; i += TPB) { s_hx[i] = 0u; s_hd[i] = 0u; }
    if (tid < 2) s_sq[tid] = 0.0f;

    // ---- one-time per-workgroup: bake B fragments + bias table ----
    const float* pch = params + (size_t)ch * NET.wtotal;
    for (int idx = tid; idx < NLAYERS * 512; idx += TPB) {
        const int L = idx >> 9;
        const int r = idx & 511;
        const int l = r >> 4;          // lane
        const int j = r & 15;          // half within lane
        const int n = l & 15;
        const int K = ((j & 8) ? 16 : 0) + ((l & 16) ? 8 : 0) + (j & 7);
        const int linW = NET.lin[L], lout = NET.lout[L];
        float v = 0.0f;
        if (n < lout && K < linW) v = pch[NET.woff[L] + n * linW + K];
        s_bfrag[L][l][j] = (_Float16)v;
    }
    for (int idx = tid; idx < NLAYERS * 16; idx += TPB) {
        const int L = idx >> 4;
        const int n = idx & 15;
        s_biasL[L][n] = (n < NET.lout[L]) ? pch[NET.boff[L] + n] : 0.0f;
    }
    __syncthreads();

    float acc_x = 0.0f, acc_d = 0.0f;

    const int row  = lane & 15;
    const int hiw  = (lane >> 4) & 1;   // which half-wave
    const int n    = lane & 15;         // output column for B/C/D

    for (int it = 0; it < TILES_WAVE; ++it) {
        const int buf = it & 1;

        // ---- wait for this tile's async pixel copy ----
        asyncWait0();

        // stats for xb + errors[0], read from LDS staging
        if (lane < 16) {
            float px[16];
            #pragma unroll
            for (int p = 0; p < 16; ++p) px[p] = s_xs[buf][wv][lane][p];
            #pragma unroll
            for (int p = 0; p < 16; ++p) {
                acc_x += px[p] * px[p];
                histAdd(s_hx, px[p]);
            }
            const float e0 = px[1] - px[0];   // errors[0] = x1 - x0
            s_e[wv][lane][0] = e0;
            acc_d += e0 * e0;
            histAdd(s_hd, e0);
        }

        // ---- issue prefetch for next tile into the other buffer ----
        if (it + 1 < TILES_WAVE) {
            const int tnext = blockIdx.y * TILES_WG + wv * TILES_WAVE + (it + 1);
            #pragma unroll
            for (int q8 = 0; q8 < 2; ++q8) {
                const int q   = q8 * 32 + lane;
                const int blk = q >> 2, r = q & 3;
                const int blkId = tnext * 16 + blk;
                const int bi = blkId >> 7, bj = blkId & 127;
                const float* g = xbase + (size_t)(bi * 4 + r) * IMG + (size_t)(bj * 4);
                const uint32_t loff = (uint32_t)(uintptr_t)&s_xs[buf ^ 1][wv][blk][r * 4];
                asyncCopy16(loff, g);
            }
        }
        __syncthreads();

        // ---- 14 sequential prediction stages ----
        for (int s = 0; s < NSTAGES; ++s) {
            const int kp = s + 2;

            // build stage input tile: [x0..x_{kp-1}, e0..e_{kp-2}], zero pad to 32
            {
                const int cb = hiw ? 16 : 0;
                #pragma unroll
                for (int j = 0; j < 16; ++j) {
                    const int col = cb + j;
                    float v = 0.0f;
                    if (col < kp)              v = s_xs[buf][wv][row][col];
                    else if (col < 2 * kp - 1) v = s_e[wv][row][col - kp];
                    s_act[wv][row][col] = (_Float16)v;
                }
            }
            __syncthreads();

            const int nl = NET.stage_count[s];
            const int lf = NET.stage_first[s];
            for (int li = 0; li < nl; ++li) {
                const int L = lf + li;

                // A fragment: two aligned 16B LDS loads + concat.
                // chunks of 8 halfs: 0:K0-7 1:K8-15 2:K16-23 3:K24-31
                const v8h* actrow = reinterpret_cast<const v8h*>(&s_act[wv][row][0]);
                const v8h alo = actrow[hiw];
                const v8h ahi = actrow[2 + hiw];
                const v16h a = __builtin_shufflevector(alo, ahi,
                    0, 1, 2, 3, 4, 5, 6, 7, 8, 9, 10, 11, 12, 13, 14, 15);

                // B fragment: two aligned 16B LDS loads from baked tile.
                const v8h* bp = reinterpret_cast<const v8h*>(&s_bfrag[L][lane][0]);
                const v8h blo = bp[0];
                const v8h bhi = bp[1];
                const v16h b = __builtin_shufflevector(blo, bhi,
                    0, 1, 2, 3, 4, 5, 6, 7, 8, 9, 10, 11, 12, 13, 14, 15);

                // C fragment: zero-padded bias broadcast down rows
                const float bias = s_biasL[L][n];
                v8f c;
                #pragma unroll
                for (int v = 0; v < 8; ++v) c[v] = bias;

                v8f d = __builtin_amdgcn_wmma_f32_16x16x32_f16(
                    /*neg_a=*/false, a, /*neg_b=*/false, b,
                    /*c_mod=*/(short)0, c, /*reuse_a=*/false, /*reuse_b=*/false);

                if (li < nl - 1) {
                    // leaky ReLU, write back as next layer's input
                    #pragma unroll
                    for (int v = 0; v < 8; ++v) {
                        float dv = d[v];
                        dv = dv > 0.0f ? dv : 0.01f * dv;
                        const int M = v + (hiw ? 8 : 0);
                        s_act[wv][M][n] = (_Float16)dv;
                    }
                    if (li == 0) {
                        // cols 16..31 held stage-input residue; zero once with
                        // one 16B vector store per lane (full row coverage).
                        v8h z = {};
                        reinterpret_cast<v8h*>(&s_act[wv][row][16])[hiw] = z;
                    }
                } else {
                    // final layer: clamp, residual, C-style fmod wrap, stats
                    if (n == 0) {
                        #pragma unroll
                        for (int v = 0; v < 8; ++v) {
                            const int M = v + (hiw ? 8 : 0);
                            float p = d[v];
                            p = p > 1.0f ? 1.0f : (p < -1.0f ? -1.0f : p);
                            const float dl = s_xs[buf][wv][M][kp] - p;
                            float w2 = dl + 1.0f;
                            w2 = w2 - truncf(w2 * 0.5f) * 2.0f;   // C-style fmod(.,2)
                            const float e = w2 - 1.0f;
                            s_e[wv][M][kp - 1] = e;
                            acc_d += e * e;
                            histAdd(s_hd, e);
                        }
                    }
                }
                __syncthreads();
            } // layers
        } // stages
        __syncthreads();
    } // tiles

    // ---- reductions: sums via LDS, histograms flushed per workgroup ----
    atomicAdd(&s_sq[0], acc_x);
    atomicAdd(&s_sq[1], acc_d);
    __syncthreads();
    if (tid == 0) {
        atomicAdd(&sums[0], s_sq[0]);
        atomicAdd(&sums[1], s_sq[1]);
    }
    for (int i = tid; i < 256; i += TPB) {
        if (s_hx[i]) atomicAdd(&hist_x[bc * 256 + i], s_hx[i]);
        if (s_hd[i]) atomicAdd(&hist_d[bc * 256 + i], s_hd[i]);
    }
}

// ---------------- finalize: entropies + RMSEs -> 4 scalars ------------------
__global__ __launch_bounds__(256)
void codec_final(const unsigned* __restrict__ hx, const unsigned* __restrict__ hd,
                 const float* __restrict__ sums, float* __restrict__ out) {
    __shared__ float ent[2];
    const int t = threadIdx.x;
    if (t < 2) ent[t] = 0.0f;
    __syncthreads();
    if (t < BC) {
        const unsigned* row = hx + t * 256;
        float e = 0.0f;
        for (int i = 0; i < 256; ++i) {
            const float p = (float)row[i] / RES_X;
            if (p > 0.0f) e -= p * log2f(p);
        }
        atomicAdd(&ent[0], e);
    } else if (t < 2 * BC) {
        const unsigned* row = hd + (t - BC) * 256;
        float e = 0.0f;
        for (int i = 0; i < 256; ++i) {
            const float p = (float)row[i] / RES_D;
            if (p > 0.0f) e -= p * log2f(p);
        }
        atomicAdd(&ent[1], e);
    }
    __syncthreads();
    if (t == 0) {
        out[0] = 255.0f * sqrtf(sums[1] / ND_ELEMS);   // loss1 (deltas)
        out[1] = 255.0f * sqrtf(sums[0] / NX_ELEMS);   // loss0 (xb)
        out[2] = ent[0] / (8.0f * (float)BC);          // invCR0
        out[3] = ent[1] / (8.0f * (float)BC);          // invCR1
    }
}

// ---------------- launch ----------------------------------------------------
extern "C" void kernel_launch(void* const* d_in, const int* in_sizes, int n_in,
                              void* d_out, int out_size, void* d_ws, size_t ws_size,
                              hipStream_t stream) {
    (void)in_sizes; (void)n_in; (void)out_size; (void)ws_size;

    const float* x = (const float*)d_in[0];

    // workspace layout: [packed params (3 * wtotal f32)] [hist_x 96*256 u32]
    //                   [hist_d 96*256 u32] [sums 2 f32]
    float* p_ws = (float*)d_ws;
    const size_t paramsBytes = (size_t)3 * NET.wtotal * sizeof(float);
    unsigned char* base = (unsigned char*)d_ws;
    unsigned* hist_x = (unsigned*)(base + paramsBytes);
    unsigned* hist_d = hist_x + (size_t)BC * 256;
    float*    sums   = (float*)(hist_d + (size_t)BC * 256);

    // zero accumulators (hist_x, hist_d, sums are contiguous)
    hipMemsetAsync(hist_x, 0, (size_t)2 * BC * 256 * sizeof(unsigned) + 2 * sizeof(float),
                   stream);

    // pack per-channel params from d_in[1..] : order is [c][kp][layer] (W, b)
    int idx = 1;
    for (int c = 0; c < CHAN; ++c) {
        for (int L = 0; L < NET.nlayers; ++L) {
            const size_t wsz = (size_t)NET.lout[L] * NET.lin[L];
            const size_t bsz = (size_t)NET.lout[L];
            hipMemcpyAsync(p_ws + (size_t)c * NET.wtotal + NET.woff[L], d_in[idx++],
                           wsz * sizeof(float), hipMemcpyDeviceToDevice, stream);
            hipMemcpyAsync(p_ws + (size_t)c * NET.wtotal + NET.boff[L], d_in[idx++],
                           bsz * sizeof(float), hipMemcpyDeviceToDevice, stream);
        }
    }

    dim3 grid(BC, CHUNKS, 1);
    codec_main<<<grid, TPB, 0, stream>>>(x, p_ws, hist_x, hist_d, sums);
    codec_final<<<1, 256, 0, stream>>>(hist_x, hist_d, sums, (float*)d_out);
}
